// SpatialAttentionModule_35596688949573
// MI455X (gfx1250) — matrix operand order repaired and checked by reading, not compiled, not run on hardware
//
#include <hip/hip_runtime.h>
#include <hip/hip_bf16.h>
#include <math.h>
#include <stdint.h>

// ---------------------------------------------------------------------------
// SpatialAttentionModule for MI455X (gfx1250, wave32, WMMA f16 16x16x32,
// Tensor Data Mover staging).
//
//   k_prep_w     : proj weights f32 -> f16  A-layout [T][Cout][Cin*9]
//   k_norm       : per-pixel L2 norm over C; emit fn (normalized) + f (raw) f16
//   k_aff_gemm   : aff[b,t] = fn^T fn   (WMMA K=256, TDM-staged tiles)
//   k_fusion     : grouped 3x3 conv over interleaved affinities -> Mf f16
//   k_value_gemm : implicit-GEMM 3x3 conv (WMMA K=2304, im2col staging)
//   k_attend     : out = 0.5*(value x attention) + 0.5*f
//                  (WMMA K=1024, double-buffered TDM staging)
// ---------------------------------------------------------------------------

typedef __attribute__((ext_vector_type(16))) _Float16 v16h;
typedef __attribute__((ext_vector_type(8)))  float    v8f;
typedef __attribute__((ext_vector_type(2)))  _Float16 h2;
typedef __attribute__((ext_vector_type(4)))  unsigned int u32x4;
typedef __attribute__((ext_vector_type(8)))  int i32x8;
typedef __attribute__((ext_vector_type(4)))  int i32x4;

#define B_   16
#define C_   256
#define HW_  1024
#define T_   2
#define KIM_ 2304   // C_ * 9

#if __has_builtin(__builtin_amdgcn_tensor_load_to_lds) && \
    __has_builtin(__builtin_amdgcn_s_wait_tensorcnt)
#define HAS_TDM 1
#else
#define HAS_TDM 0
#endif

#if HAS_TDM
// ---- TDM 2-D f16 tile load: global[rows x cols, row stride] -> LDS row-major
// D# per ISA cdna5 ch.8: group0 {count, lds_addr, global_addr, type=2},
// group1 {data_size=2B, tensor dims == tile dims, tile dims, dim0 stride}.
// This toolchain exposes the 6-arg builtin (g0, g1, g2, g3, g4, cpol).
__device__ __forceinline__ void tdm_load_tile_f16(const _Float16* gsrc,
                                                  _Float16* lds_dst,
                                                  int rows, int cols,
                                                  long long row_stride_elems) {
  unsigned long long ga = (unsigned long long)(uintptr_t)gsrc;
  unsigned lds = (unsigned)(uintptr_t)lds_dst;   // LDS aperture: addr[31:0]
  u32x4 g0;
  g0.x = 1u;                                           // count=1, user mode
  g0.y = lds;                                          // lds_addr
  g0.z = (unsigned)ga;                                 // global_addr[31:0]
  g0.w = (unsigned)((ga >> 32) & 0x01ffffffull) | (2u << 30);  // ga[56:32]|type=2
  unsigned long long s0 = (unsigned long long)row_stride_elems;
  unsigned r = (unsigned)rows, c = (unsigned)cols;
  i32x8 g1;
  g1[0] = (int)(1u << 16);                             // data_size=1 (2 bytes)
  g1[1] = (int)((c & 0xffffu) << 16);                  // tensor_dim0[15:0]
  g1[2] = (int)(((c >> 16) & 0xffffu) | ((r & 0xffffu) << 16));  // d0 hi|d1 lo
  g1[3] = (int)(((r >> 16) & 0xffffu) | ((c & 0xffffu) << 16));  // d1 hi|tile_d0
  g1[4] = (int)(r & 0xffffu);                          // tile_dim1 (tile_dim2=0)
  g1[5] = (int)(unsigned)(s0 & 0xffffffffull);         // dim0_stride lo
  g1[6] = (int)(unsigned)((s0 >> 32) & 0xffffull);     // dim0_stride hi
  g1[7] = 0;
  i32x4 gz4 = {0, 0, 0, 0};
  i32x8 gz8 = {0, 0, 0, 0, 0, 0, 0, 0};
  __builtin_amdgcn_tensor_load_to_lds(g0, g1, gz4, gz4, gz8, 0);
}
#endif

// ---- fragment builders (ISA 7.12.2 16-bit layouts, wave32) -----------------

// A fragment 16x32 f16 from a [*, ldk] row-major (K-contiguous) array.
__device__ __forceinline__ v16h frag_a(const _Float16* Abase, int ldk,
                                       int m0, int k0, int lane) {
  const int row = lane & 15;
  const int g8  = (lane >> 4) << 3;     // 0 or 8
  const _Float16* r = Abase + (size_t)(m0 + row) * ldk + k0;
  v16h a;
#pragma unroll
  for (int i = 0; i < 4; ++i) {
    h2 p0 = *(const h2*)&r[g8 + 2 * i];
    h2 p1 = *(const h2*)&r[16 + g8 + 2 * i];
    a[2 * i]     = p0.x;  a[2 * i + 1]     = p0.y;
    a[8 + 2 * i] = p1.x;  a[8 + 2 * i + 1] = p1.y;
  }
  return a;
}

// B fragment 32x16 f16 from an LDS tile stored [n][32] (n-major, K-contiguous).
__device__ __forceinline__ v16h frag_b(const _Float16* Bs, int nsub, int lane) {
  const int col = lane & 15;
  const int g8  = (lane >> 4) << 3;     // 0 or 8
  const _Float16* r = Bs + (size_t)(nsub * 16 + col) * 32;
  v16h b;
#pragma unroll
  for (int i = 0; i < 4; ++i) {
    h2 p0 = *(const h2*)&r[g8 + 2 * i];
    h2 p1 = *(const h2*)&r[16 + g8 + 2 * i];
    b[2 * i] = p0.x;  b[2 * i + 1] = p0.y;
    b[8 + 2 * i] = p1.x;  b[8 + 2 * i + 1] = p1.y;
  }
  return b;
}

// A fragment from a K-major LDS tile S[32 k][ldp m] (TDM natural order).
__device__ __forceinline__ v16h frag_a_t(const _Float16* S, int ldp, int lane) {
  const int row = lane & 15;
  const int g8  = (lane >> 4) << 3;
  v16h a;
#pragma unroll
  for (int i = 0; i < 8; ++i) {
    a[i]     = S[(g8 + i) * ldp + row];
    a[8 + i] = S[(16 + g8 + i) * ldp + row];
  }
  return a;
}

// B fragment from a K-major LDS tile S[32 k][ldp n].
__device__ __forceinline__ v16h frag_b_t(const _Float16* S, int ldp, int nsub,
                                         int lane) {
  const int col = (lane & 15) + nsub * 16;
  const int g8  = (lane >> 4) << 3;
  v16h b;
#pragma unroll
  for (int i = 0; i < 8; ++i) {
    b[i]     = S[(g8 + i) * ldp + col];
    b[8 + i] = S[(16 + g8 + i) * ldp + col];
  }
  return b;
}

__device__ __forceinline__ v8f wmma_f16(v16h a, v16h b, v8f c) {
  return __builtin_amdgcn_wmma_f32_16x16x32_f16(
      false, a, false, b, (short)0, c, false, false);
}

// ---- k_prep_w --------------------------------------------------------------
__global__ void k_prep_w(const float* __restrict__ wseg,
                         const float* __restrict__ wdep,
                         _Float16* __restrict__ Wh) {
  int gid = blockIdx.x * blockDim.x + threadIdx.x;
  const int total = T_ * C_ * KIM_;
  if (gid >= total) return;
  int t = gid / (C_ * KIM_);
  int r = gid - t * (C_ * KIM_);
  const float* w = t ? wdep : wseg;
  Wh[gid] = (_Float16)w[r];
}

// ---- k_norm ----------------------------------------------------------------
__global__ void k_norm(const float* __restrict__ fseg,
                       const float* __restrict__ fdep,
                       _Float16* __restrict__ fnh,
                       _Float16* __restrict__ fh) {
  int gid = blockIdx.x * blockDim.x + threadIdx.x;  // over B*T*HW
  int p  = gid & (HW_ - 1);
  int bt = gid >> 10;
  int t  = bt & 1;
  int b  = bt >> 1;
  const float* f = (t ? fdep : fseg) + (size_t)b * C_ * HW_ + p;
  float ss = 0.f;
  for (int c = 0; c < C_; ++c) {
    float v = f[(size_t)c * HW_];
    ss += v * v;
  }
  float inv = 1.f / fmaxf(sqrtf(ss), 1e-12f);
  _Float16* on = fnh + (size_t)bt * C_ * HW_ + p;
  _Float16* oh = fh  + (size_t)bt * C_ * HW_ + p;
  for (int c = 0; c < C_; ++c) {
    float v = f[(size_t)c * HW_];
    on[(size_t)c * HW_] = (_Float16)(v * inv);
    oh[(size_t)c * HW_] = (_Float16)v;
  }
}

// ---- k_aff_gemm: aff[b,t] = fn^T * fn  (M=N=HW, K=C) -----------------------
__global__ void __launch_bounds__(32)
k_aff_gemm(const _Float16* __restrict__ fnh, _Float16* __restrict__ aff) {
  const int nt = blockIdx.x;   // HW/64
  const int mt = blockIdx.y;   // HW/16
  const int bt = blockIdx.z;   // B*T
  const int lane = threadIdx.x;
  const _Float16* F = fnh + (size_t)bt * C_ * HW_;   // [C][HW]
  const int m0 = mt * 16, n0 = nt * 64;
  v8f acc[4] = {};
#if HAS_TDM
  __shared__ _Float16 Sa[32 * 16];   // [k(c)][m]  TDM natural order
  __shared__ _Float16 Sb[32 * 64];   // [k(c)][n]
  for (int k0 = 0; k0 < C_; k0 += 32) {
    __syncthreads();   // prior ds reads done before TDM overwrites tiles
    tdm_load_tile_f16(F + (size_t)k0 * HW_ + m0, Sa, 32, 16, HW_);
    tdm_load_tile_f16(F + (size_t)k0 * HW_ + n0, Sb, 32, 64, HW_);
    __builtin_amdgcn_s_wait_tensorcnt(0);
    v16h a = frag_a_t(Sa, 16, lane);
#pragma unroll
    for (int s = 0; s < 4; ++s)
      acc[s] = wmma_f16(a, frag_b_t(Sb, 64, s, lane), acc[s]);
  }
#else
  __shared__ _Float16 As[16 * 32];
  __shared__ _Float16 Bs[64 * 32];
  for (int k0 = 0; k0 < C_; k0 += 32) {
    const _Float16* rowc = F + (size_t)(k0 + lane) * HW_;
#pragma unroll
    for (int m = 0; m < 16; ++m) As[m * 32 + lane] = rowc[m0 + m];
#pragma unroll
    for (int n = 0; n < 64; ++n) Bs[n * 32 + lane] = rowc[n0 + n];
    __syncthreads();
    v16h a = frag_a(As, 32, 0, 0, lane);
#pragma unroll
    for (int s = 0; s < 4; ++s)
      acc[s] = wmma_f16(a, frag_b(Bs, s, lane), acc[s]);
    __syncthreads();
  }
#endif
  _Float16* O = aff + (size_t)bt * HW_ * HW_;
  const int nn = lane & 15, mb = (lane >> 4) * 8;
#pragma unroll
  for (int s = 0; s < 4; ++s)
#pragma unroll
    for (int r = 0; r < 8; ++r)
      O[(size_t)(m0 + mb + r) * HW_ + (n0 + s * 16 + nn)] = (_Float16)acc[s][r];
}

// ---- k_fusion: grouped conv (groups=HW, cin=T, 3x3, SAME) -------------------
__global__ void k_fusion(const _Float16* __restrict__ aff,
                         const float* __restrict__ fw,
                         const float* __restrict__ fb,
                         _Float16* __restrict__ Mf) {
  long gid = (long)blockIdx.x * blockDim.x + threadIdx.x;
  if (gid >= (long)B_ * HW_ * HW_) return;
  int j = (int)(gid & (HW_ - 1));
  int i = (int)((gid >> 10) & (HW_ - 1));
  int b = (int)(gid >> 20);
  int y = j >> 5, x = j & 31;
  float acc = fb[i];
#pragma unroll
  for (int t = 0; t < T_; ++t) {
    const _Float16* a = aff + ((size_t)(b * T_ + t) * HW_ + i) * HW_;
    const float* w = fw + ((size_t)i * T_ + t) * 9;
#pragma unroll
    for (int ky = 0; ky < 3; ++ky) {
      int yy = y + ky - 1;
      if ((unsigned)yy >= 32u) continue;
#pragma unroll
      for (int kx = 0; kx < 3; ++kx) {
        int xx = x + kx - 1;
        if ((unsigned)xx >= 32u) continue;
        acc += w[ky * 3 + kx] * (float)a[yy * 32 + xx];
      }
    }
  }
  Mf[((size_t)b * HW_ + i) * HW_ + j] = (_Float16)acc;
}

// ---- k_value_gemm: implicit-GEMM 3x3 conv, M=C, N=HW, K=2304 ---------------
__global__ void __launch_bounds__(32)
k_value_gemm(const _Float16* __restrict__ Wh,
             const _Float16* __restrict__ fh,
             const float* __restrict__ bseg,
             const float* __restrict__ bdep,
             _Float16* __restrict__ valh) {
  const int nt = blockIdx.x;   // HW/64
  const int mt = blockIdx.y;   // C/16
  const int bt = blockIdx.z;
  const int t  = bt & 1;
  const int lane = threadIdx.x;
  const _Float16* A   = Wh + (size_t)t * C_ * KIM_;
  const _Float16* Fin = fh + (size_t)bt * C_ * HW_;
  __shared__ _Float16 Bs[64 * 32];
  const int m0 = mt * 16, n0 = nt * 64;
  const int y0 = n0 >> 5;
  v8f acc[4] = {};
  for (int k0 = 0; k0 < KIM_; k0 += 32) {
    int k  = k0 + lane;
    int ci = k / 9, tap = k - ci * 9;
    int dy = tap / 3 - 1, dx = tap - (tap / 3) * 3 - 1;
    const _Float16* src = Fin + (size_t)ci * HW_;
#pragma unroll
    for (int n = 0; n < 64; ++n) {
      int yy = y0 + (n >> 5) + dy;
      int xx = (n & 31) + dx;
      _Float16 v = (_Float16)0.f;
      if ((unsigned)yy < 32u && (unsigned)xx < 32u) v = src[yy * 32 + xx];
      Bs[n * 32 + lane] = v;
    }
    __syncthreads();
    if (k0 + 32 < KIM_)
      __builtin_prefetch(A + (size_t)m0 * KIM_ + k0 + 32, 0, 1);
    v16h a = frag_a(A, KIM_, m0, k0, lane);
#pragma unroll
    for (int s = 0; s < 4; ++s)
      acc[s] = wmma_f16(a, frag_b(Bs, s, lane), acc[s]);
    __syncthreads();
  }
  const float* bias = t ? bdep : bseg;
  _Float16* O = valh + (size_t)bt * C_ * HW_;
  const int nn = lane & 15, mb = (lane >> 4) * 8;
#pragma unroll
  for (int s = 0; s < 4; ++s)
#pragma unroll
    for (int r = 0; r < 8; ++r) {
      float v = acc[s][r] + bias[m0 + mb + r];
      O[(size_t)(m0 + mb + r) * HW_ + (n0 + s * 16 + nn)] = (_Float16)v;
    }
}

// ---- k_attend: out = 0.5*(value x attention) + 0.5*f -----------------------
__global__ void __launch_bounds__(32)
k_attend(const _Float16* __restrict__ valh,
         const _Float16* __restrict__ Mf,
         const float* __restrict__ fseg,
         const float* __restrict__ fdep,
         float* __restrict__ dout) {
  const int nt = blockIdx.x;   // HW/64
  const int mt = blockIdx.y;   // C/16
  const int bt = blockIdx.z;
  const int b = bt >> 1, t = bt & 1;
  const int lane = threadIdx.x;
  const _Float16* A  = valh + (size_t)bt * C_ * HW_;
  const _Float16* Bm = Mf + (size_t)b * HW_ * HW_;
  const int m0 = mt * 16, n0 = nt * 64;
  v8f acc[4] = {};
#if HAS_TDM
  // Double-buffered TDM staging: attention tile is K-contiguous in Mf, so the
  // DMA lands it directly in fragment-friendly [n][32] order.
  __shared__ _Float16 Bsb[2][64 * 32];
  tdm_load_tile_f16(Bm + (size_t)n0 * HW_, Bsb[0], 64, 32, HW_);
  for (int k0 = 0; k0 < HW_; k0 += 32) {
    const int cur = (k0 >> 5) & 1;
    __syncthreads();   // reads of Bsb[cur^1] (from iter-1) done before refill
    if (k0 + 32 < HW_) {
      tdm_load_tile_f16(Bm + (size_t)n0 * HW_ + (k0 + 32), Bsb[cur ^ 1],
                        64, 32, HW_);
      __builtin_amdgcn_s_wait_tensorcnt(1);   // tile `cur` landed
    } else {
      __builtin_amdgcn_s_wait_tensorcnt(0);
    }
    if (k0 + 32 < HW_)
      __builtin_prefetch(A + (size_t)m0 * HW_ + k0 + 32, 0, 1);
    v16h a = frag_a(A, HW_, m0, k0, lane);
#pragma unroll
    for (int s = 0; s < 4; ++s)
      acc[s] = wmma_f16(a, frag_b(&Bsb[cur][0], s, lane), acc[s]);
  }
#else
  __shared__ _Float16 Bs[64 * 32];
  for (int k0 = 0; k0 < HW_; k0 += 32) {
#pragma unroll
    for (int rr = 0; rr < 2; ++rr) {
      int n = lane + rr * 32;
      const _Float16* src = Bm + (size_t)(n0 + n) * HW_ + k0;
      _Float16* dst = &Bs[n * 32];
#pragma unroll
      for (int jj = 0; jj < 16; ++jj)
        *(h2*)&dst[2 * jj] = *(const h2*)&src[2 * jj];
    }
    __syncthreads();
    v16h a = frag_a(A, HW_, m0, k0, lane);
#pragma unroll
    for (int s = 0; s < 4; ++s)
      acc[s] = wmma_f16(a, frag_b(Bs, s, lane), acc[s]);
    __syncthreads();
  }
#endif
  const float* forig = t ? fdep : fseg;
  float* O = dout + (size_t)t * B_ * C_ * HW_;
  const int nn = lane & 15, mb = (lane >> 4) * 8;
#pragma unroll
  for (int s = 0; s < 4; ++s)
#pragma unroll
    for (int r = 0; r < 8; ++r) {
      int c = m0 + mb + r, p = n0 + s * 16 + nn;
      size_t idx = ((size_t)b * C_ + c) * HW_ + p;
      O[idx] = 0.5f * acc[s][r] + 0.5f * forig[idx];
    }
}

// ---------------------------------------------------------------------------
extern "C" void kernel_launch(void* const* d_in, const int* in_sizes, int n_in,
                              void* d_out, int out_size, void* d_ws, size_t ws_size,
                              hipStream_t stream) {
  (void)in_sizes; (void)n_in; (void)out_size; (void)ws_size;
  const float* fseg = (const float*)d_in[0];
  const float* fdep = (const float*)d_in[1];
  const float* wseg = (const float*)d_in[2];
  const float* bseg = (const float*)d_in[3];
  const float* wdep = (const float*)d_in[4];
  const float* bdep = (const float*)d_in[5];
  const float* fw   = (const float*)d_in[6];
  const float* fb   = (const float*)d_in[7];
  float* out = (float*)d_out;

  char* ws = (char*)d_ws;
  size_t off = 0;
  _Float16* fnh  = (_Float16*)(ws + off); off += (size_t)B_ * T_ * C_ * HW_ * 2;
  _Float16* fh   = (_Float16*)(ws + off); off += (size_t)B_ * T_ * C_ * HW_ * 2;
  _Float16* Wh   = (_Float16*)(ws + off); off += (size_t)T_ * C_ * KIM_ * 2;
  off = (off + 255) & ~(size_t)255;
  _Float16* aff  = (_Float16*)(ws + off); off += (size_t)B_ * T_ * HW_ * HW_ * 2;
  _Float16* Mf   = (_Float16*)(ws + off); off += (size_t)B_ * HW_ * HW_ * 2;
  _Float16* valh = (_Float16*)(ws + off);

  k_prep_w<<<dim3((T_ * C_ * KIM_ + 255) / 256), 256, 0, stream>>>(wseg, wdep, Wh);
  k_norm<<<dim3((B_ * T_ * HW_) / 256), 256, 0, stream>>>(fseg, fdep, fnh, fh);
  k_aff_gemm<<<dim3(HW_ / 64, HW_ / 16, B_ * T_), 32, 0, stream>>>(fnh, aff);
  k_fusion<<<dim3((unsigned)(((long)B_ * HW_ * HW_ + 255) / 256)), 256, 0, stream>>>(
      aff, fw, fb, Mf);
  k_value_gemm<<<dim3(HW_ / 64, C_ / 16, B_ * T_), 32, 0, stream>>>(
      Wh, fh, bseg, bdep, valh);
  k_attend<<<dim3(HW_ / 64, C_ / 16, B_ * T_), 32, 0, stream>>>(
      valh, Mf, fseg, fdep, out);
}
